// GumbelWav2Vec2VectorQuantizer_63745904608035
// MI455X (gfx1250) — matrix operand order repaired and checked by reading, not compiled, not run on hardware
//
#include <hip/hip_runtime.h>
#include <hip/hip_bf16.h>

// ---------------------------------------------------------------------------
// GumbelWav2Vec2VectorQuantizer fused kernel for MI455X (gfx1250, wave32)
//
// logits = x @ W^T + b           (M=48000, N=640, K=512)  -> bf16x3 WMMA
// per (row, group of 320):
//   argmax(logits)                -> hard one-hot counts (code perplexity)
//   softmax(logits)               -> avg prob accumulation (prob perplexity)
//   argmax(logits + gumbel_noise) -> k_g ; cb = one-hot(k_g)
//   quantized = entries[g, k_g, :]
// ---------------------------------------------------------------------------

typedef __attribute__((ext_vector_type(16))) __bf16 v16bf;
typedef __attribute__((ext_vector_type(8)))  __bf16 v8bf;
typedef __attribute__((ext_vector_type(8)))  float  v8f;

#define BSZ     32
#define TSZ     1500
#define NROWS   48000      // BSZ*TSZ
#define KDIM    512
#define GE      640        // G*E
#define GRP     2
#define ENT     320        // E
#define DDIM    384        // D
#define OUTD    768        // G*D
#define MROWS   64         // rows per workgroup
#define NBLK    750        // NROWS / MROWS
#define THREADS 512        // 16 waves
#define EPS_P   1e-7f
#define EPS_G   1e-10f

__device__ __forceinline__ unsigned short f2bf_rne(float x) {
    unsigned int u = __float_as_uint(x);
    unsigned int r = u + 0x7FFFu + ((u >> 16) & 1u);
    return (unsigned short)(r >> 16);
}
__device__ __forceinline__ float bf2f(unsigned short h) {
    return __uint_as_float(((unsigned int)h) << 16);
}

// ---------------------------------------------------------------------------
// Zero the global stat accumulators: [0..639] soft-prob sums, [640..1279] hard counts
__global__ void vq_zero(float* __restrict__ gAcc) {
    int i = blockIdx.x * blockDim.x + threadIdx.x;
    if (i < 2 * GE) gAcc[i] = 0.0f;
}

// ---------------------------------------------------------------------------
// Split W (fp32, [640,512] row-major) into bf16 hi/lo planes (bf16x3 scheme).
__global__ void vq_split_w(const float* __restrict__ W,
                           unsigned short* __restrict__ Whi,
                           unsigned short* __restrict__ Wlo) {
    int i = blockIdx.x * blockDim.x + threadIdx.x;
    if (i < GE * KDIM) {
        float w = W[i];
        unsigned short h = f2bf_rne(w);
        Whi[i] = h;
        Wlo[i] = f2bf_rne(w - bf2f(h));
    }
}

// ---------------------------------------------------------------------------
// Fused GEMM + gumbel VQ. Grid: 750 blocks x 512 threads (16 waves).
// Wave w: mtile = w>>2 (16-row slab), nquad = w&3 (160-col slab, 10 N-tiles).
__global__ void __launch_bounds__(THREADS)
vq_main(const float* __restrict__ x,
        const float* __restrict__ bias,
        const float* __restrict__ entries,
        const float* __restrict__ gum,
        const unsigned short* __restrict__ Whi,
        const unsigned short* __restrict__ Wlo,
        float* __restrict__ qOut,
        float* __restrict__ cbOut,
        float* __restrict__ gAcc) {
    // Overlaid LDS: phase 1 = A tile bf16 hi/lo (64x512 each, 128KB total),
    //               phase 2 = logits tile f32 (64x640, 160KB).
    __shared__ __align__(16) unsigned char smemRaw[MROWS * GE * 4]; // 160KB
    __shared__ float sAcc[2 * GE];                                   // 5KB

    unsigned short* sAhi = (unsigned short*)smemRaw;            // [64*512]
    unsigned short* sAlo = sAhi + MROWS * KDIM;                 // [64*512]
    float* sLogits = (float*)smemRaw;                           // [64*640] (overlay)

    const int tid  = threadIdx.x;
    const int lane = tid & 31;
    const int wave = tid >> 5;
    const size_t n0 = (size_t)blockIdx.x * MROWS;

    for (int i = tid; i < 2 * GE; i += THREADS) sAcc[i] = 0.0f;

    // ---- Stage 1: load x tile (64x512 f32) and split to bf16 hi/lo in LDS
    {
        const float4* xv = (const float4*)(x + n0 * KDIM); // 8192 float4
        for (int i = tid; i < (MROWS * KDIM) / 4; i += THREADS) {
            float4 v = xv[i];
            int base = i * 4;
            float vv[4] = {v.x, v.y, v.z, v.w};
            #pragma unroll
            for (int j = 0; j < 4; ++j) {
                unsigned short h = f2bf_rne(vv[j]);
                sAhi[base + j] = h;
                sAlo[base + j] = f2bf_rne(vv[j] - bf2f(h));
            }
        }
    }
    __syncthreads();

    // ---- Stage 2: bf16x3 WMMA GEMM into fp32 accumulators
    const int mtile   = wave >> 2;        // 0..3
    const int nquad   = wave & 3;         // 0..3
    const int colLane = lane & 15;
    const int rowA    = mtile * 16 + colLane;
    const int kg8     = (lane >> 4) * 8;  // unused; A loads also use contiguous kg16
    const int kg16    = (lane >> 4) * 16; // B-fragment contiguous K halves

    v8f acc[10] = {};

    for (int k0 = 0; k0 < KDIM; k0 += 32) {
        v8bf ah0 = *(const v8bf*)(sAhi + rowA * KDIM + k0 + kg16);
        v8bf ah1 = *(const v8bf*)(sAhi + rowA * KDIM + k0 + kg16 + 8);
        v8bf al0 = *(const v8bf*)(sAlo + rowA * KDIM + k0 + kg16);
        v8bf al1 = *(const v8bf*)(sAlo + rowA * KDIM + k0 + kg16 + 8);
        v16bf ahi, alo;
        #pragma unroll
        for (int j = 0; j < 8; ++j) {
            ahi[j] = ah0[j]; ahi[j + 8] = ah1[j];
            alo[j] = al0[j]; alo[j + 8] = al1[j];
        }
        #pragma unroll
        for (int t = 0; t < 10; ++t) {
            const int col = (nquad * 10 + t) * 16 + colLane;
            const v16bf bhi = *(const v16bf*)(Whi + (size_t)col * KDIM + k0 + kg16);
            const v16bf blo = *(const v16bf*)(Wlo + (size_t)col * KDIM + k0 + kg16);
            acc[t] = __builtin_amdgcn_wmma_f32_16x16x32_bf16(
                false, ahi, false, bhi, (short)0, acc[t], false, false);
            acc[t] = __builtin_amdgcn_wmma_f32_16x16x32_bf16(
                false, alo, false, bhi, (short)0, acc[t], false, false);
            acc[t] = __builtin_amdgcn_wmma_f32_16x16x32_bf16(
                false, ahi, false, blo, (short)0, acc[t], false, false);
        }
    }
    __syncthreads(); // all LDS A reads complete before overlaying logits

    // ---- Stage 3: C fragments (+bias) -> LDS logits tile
    {
        const int mHalf = (lane >> 4) * 8; // C layout: VGPR r -> M=r (lanes 0-15), r+8 (16-31)
        for (int t = 0; t < 10; ++t) {
            const int col = (nquad * 10 + t) * 16 + colLane;
            const float bb = bias[col];
            #pragma unroll
            for (int r = 0; r < 8; ++r) {
                const int m = mtile * 16 + r + mHalf;
                sLogits[m * GE + col] = acc[t][r] + bb;
            }
        }
    }
    __syncthreads();

    // ---- Stage 4: per-(row,group) argmax / softmax / gumbel argmax
    for (int p = wave; p < MROWS * GRP; p += 16) {
        const int row  = p >> 1;
        const int gidx = p & 1;
        const size_t n = n0 + row;
        const float* lrow = sLogits + row * GE + gidx * ENT;
        const float* grow = gum + ((size_t)n * GRP + gidx) * ENT;

        float v[10];
        float m = -3.402823466e38f;  int mi  = 0;
        float gm = -3.402823466e38f; int gmi = 0;
        #pragma unroll
        for (int j = 0; j < 10; ++j) {
            const int e = lane + j * 32;
            const float lv = lrow[e];
            v[j] = lv;
            if (lv > m) { m = lv; mi = e; }
            const float u  = grow[e];
            const float gn = -logf(-logf(u + EPS_G) + EPS_G);
            const float gv = lv + gn;          // argmax invariant under /TAU + softmax
            if (gv > gm) { gm = gv; gmi = e; }
        }
        // wave32 argmax reductions (first-index tie-break like jnp.argmax)
        #pragma unroll
        for (int off = 16; off > 0; off >>= 1) {
            float om = __shfl_xor(m, off, 32);  int oi = __shfl_xor(mi, off, 32);
            if (om > m || (om == m && oi < mi)) { m = om; mi = oi; }
            float og = __shfl_xor(gm, off, 32); int ogi = __shfl_xor(gmi, off, 32);
            if (og > gm || (og == gm && ogi < gmi)) { gm = og; gmi = ogi; }
        }
        // softmax of plain logits -> avg_probs accumulation
        float pexp[10];
        float ssum = 0.0f;
        #pragma unroll
        for (int j = 0; j < 10; ++j) { pexp[j] = expf(v[j] - m); ssum += pexp[j]; }
        #pragma unroll
        for (int off = 16; off > 0; off >>= 1) ssum += __shfl_xor(ssum, off, 32);
        const float inv = 1.0f / ssum;
        #pragma unroll
        for (int j = 0; j < 10; ++j)
            atomicAdd(&sAcc[gidx * ENT + lane + j * 32], pexp[j] * inv);
        if (lane == 0) atomicAdd(&sAcc[GE + gidx * ENT + mi], 1.0f);

        // cb = one-hot(gumbel argmax)
        float* cbrow = cbOut + (size_t)n * GE + gidx * ENT;
        #pragma unroll
        for (int j = 0; j < 10; ++j) {
            const int e = lane + j * 32;
            cbrow[e] = (e == gmi) ? 1.0f : 0.0f;
        }
        // quantized = entries[g, k_g, :]
        const float* ent = entries + ((size_t)(gidx * ENT + gmi)) * DDIM;
        float* qrow = qOut + (size_t)n * OUTD + gidx * DDIM;
        for (int d = lane; d < DDIM; d += 32) qrow[d] = ent[d];
    }
    __syncthreads();

    // ---- Stage 5: flush per-block stats to global accumulators
    for (int i = tid; i < 2 * GE; i += THREADS) atomicAdd(&gAcc[i], sAcc[i]);
}

// ---------------------------------------------------------------------------
// Finalize: perplexities from the [2,320] soft/hard accumulators.
__global__ void vq_finalize(const float* __restrict__ gAcc,
                            float* __restrict__ outScal) {
    __shared__ float red[2 * GE];
    __shared__ float out4[4];
    const int t = threadIdx.x; // 640 threads
    const float invN = 1.0f / (float)NROWS;
    const float ps = gAcc[t]      * invN; // soft avg prob
    const float ph = gAcc[GE + t] * invN; // hard prob
    red[t]      = ps * logf(ps + EPS_P);
    red[GE + t] = ph * logf(ph + EPS_P);
    __syncthreads();
    if (t < 4) { // 0: soft g0, 1: soft g1, 2: hard g0, 3: hard g1
        const float* bp = red + (t >> 1) * GE + (t & 1) * ENT;
        float s = 0.0f;
        for (int e = 0; e < ENT; ++e) s += bp[e];
        out4[t] = s;
    }
    __syncthreads();
    if (t == 0) {
        outScal[0] = expf(-out4[2]) + expf(-out4[3]); // code_perplexity
        outScal[1] = expf(-out4[0]) + expf(-out4[1]); // prob_perplexity
    }
}

// ---------------------------------------------------------------------------
extern "C" void kernel_launch(void* const* d_in, const int* in_sizes, int n_in,
                              void* d_out, int out_size, void* d_ws, size_t ws_size,
                              hipStream_t stream) {
    const float* x       = (const float*)d_in[0]; // [32,1500,512]
    const float* W       = (const float*)d_in[1]; // [640,512]
    const float* b       = (const float*)d_in[2]; // [640]
    const float* entries = (const float*)d_in[3]; // [1,640,384]
    const float* gum     = (const float*)d_in[4]; // [96000,320]

    float* out = (float*)d_out;
    const size_t QELEMS  = (size_t)NROWS * OUTD; // 36,864,000
    const size_t CBELEMS = (size_t)NROWS * GE;   // 30,720,000
    float* qOut  = out;
    float* cbOut = out + QELEMS;
    float* scal  = out + QELEMS + CBELEMS;

    unsigned short* Whi = (unsigned short*)d_ws;
    unsigned short* Wlo = Whi + (size_t)GE * KDIM;
    float* gAcc = (float*)(Wlo + (size_t)GE * KDIM); // 1280 floats

    vq_zero<<<5, 256, 0, stream>>>(gAcc);
    vq_split_w<<<(GE * KDIM + 255) / 256, 256, 0, stream>>>(W, Whi, Wlo);
    vq_main<<<NBLK, THREADS, 0, stream>>>(x, b, entries, gum, Whi, Wlo,
                                          qOut, cbOut, gAcc);
    vq_finalize<<<1, GE, 0, stream>>>(gAcc, scal);
}
